// GDTLayer_61564061221139
// MI455X (gfx1250) — matrix operand order repaired and checked by reading, not compiled, not run on hardware
//
#include <hip/hip_runtime.h>

#define NN 50000
#define EE 640000
#define F  128
#define HH 8
#define DD 16
#define FFH 512
#define NHOP 5
#define ALPHA 0.1f
#define SLOPE 0.2f
#define LNEPS 1e-5f

typedef __attribute__((ext_vector_type(16))) __bf16 v16bf;
typedef __attribute__((ext_vector_type(8)))  float  v8f;
typedef __attribute__((ext_vector_type(4)))  unsigned int u32x4;

union ABReg { v16bf v; u32x4 q[2]; };

__device__ __forceinline__ unsigned short f2bf(float f) {
  unsigned u = __float_as_uint(f);
  u += 0x7fffu + ((u >> 16) & 1u);   // round-to-nearest-even
  return (unsigned short)(u >> 16);
}
__device__ __forceinline__ unsigned fkey(float f) {   // order-preserving map f32 -> u32
  unsigned u = __float_as_uint(f);
  return (u & 0x80000000u) ? ~u : (u | 0x80000000u);
}
__device__ __forceinline__ float fdec(unsigned k) {
  unsigned u = (k & 0x80000000u) ? (k ^ 0x80000000u) : ~k;
  return __uint_as_float(u);
}
__device__ __forceinline__ float lrelu(float p) { return p > 0.f ? p : SLOPE * p; }

// ---------------- utility kernels ----------------
__global__ void k_fill(float* p, int n, float v) {
  int i = blockIdx.x * blockDim.x + threadIdx.x;
  int st = gridDim.x * blockDim.x;
  for (; i < n; i += st) p[i] = v;
}

__global__ void k_cvt_bf16(const float* __restrict__ in, unsigned short* __restrict__ out, int n) {
  int i = blockIdx.x * blockDim.x + threadIdx.x;
  int st = gridDim.x * blockDim.x;
  for (; i < n; i += st) out[i] = f2bf(in[i]);
}

__global__ void k_deg(const int* __restrict__ src, const int* __restrict__ dst,
                      float* odeg, float* ideg) {
  int e = blockIdx.x * blockDim.x + threadIdx.x;
  if (e >= EE) return;
  atomicAdd(odeg + src[e], 1.0f);
  atomicAdd(ideg + dst[e], 1.0f);
}

__global__ void k_norms(const float* __restrict__ odeg, const float* __restrict__ ideg,
                        float* hn, float* tn, float* li) {
  int i = blockIdx.x * blockDim.x + threadIdx.x;
  if (i >= NN) return;
  float od = fmaxf(odeg[i], 1.0f);
  float id = fmaxf(ideg[i], 1.0f);
  hn[i] = rsqrtf(od);
  tn[i] = sqrtf(id);
  li[i] = log1pf(id);
}

// ---------------- LayerNorm (1 wave per row, 4 elems/lane) ----------------
__global__ void k_ln_cast(const float* __restrict__ x, const float* __restrict__ g,
                          const float* __restrict__ b, unsigned short* __restrict__ xbf) {
  int wave = threadIdx.x >> 5, lane = threadIdx.x & 31;
  int row = blockIdx.x * 8 + wave;
  float4 v = ((const float4*)(x + row * F))[lane];
  float s = v.x + v.y + v.z + v.w;
  for (int o = 16; o > 0; o >>= 1) s += __shfl_xor(s, o, 32);
  float mu = s * (1.0f / F);
  float dx = v.x - mu, dy = v.y - mu, dz = v.z - mu, dw = v.w - mu;
  float q = dx * dx + dy * dy + dz * dz + dw * dw;
  for (int o = 16; o > 0; o >>= 1) q += __shfl_xor(q, o, 32);
  float rs = rsqrtf(q * (1.0f / F) + LNEPS);
  float4 g4 = ((const float4*)g)[lane];
  float4 b4 = ((const float4*)b)[lane];
  float y0 = dx * rs * g4.x + b4.x;
  float y1 = dy * rs * g4.y + b4.y;
  float y2 = dz * rs * g4.z + b4.z;
  float y3 = dw * rs * g4.w + b4.w;
  uint2 pk;
  pk.x = (unsigned)f2bf(y0) | ((unsigned)f2bf(y1) << 16);
  pk.y = (unsigned)f2bf(y2) | ((unsigned)f2bf(y3) << 16);
  *(uint2*)(xbf + row * F + lane * 4) = pk;
}

// ---------------- fused triple projection GEMM (WMMA bf16) ----------------
// fh/ft/fe[n,o] = head_norm[n] * sum_k LN(x)[n,k] * W[o,k]
__global__ void __launch_bounds__(256) k_gemm3(
    const unsigned short* __restrict__ xbf,
    const unsigned short* __restrict__ wh, const unsigned short* __restrict__ wt,
    const unsigned short* __restrict__ we, const float* __restrict__ hn,
    float* __restrict__ fh, float* __restrict__ ft, float* __restrict__ fe) {
  int wave = threadIdx.x >> 5, lane = threadIdx.x & 31;
  int l15 = lane & 15, asel = lane >> 4;
  int m0 = blockIdx.x * 16;
  int colbase = wave * 16;
  v8f a0 = {0,0,0,0,0,0,0,0}, a1 = a0, a2 = a0;
#pragma unroll
  for (int ks = 0; ks < 4; ks++) {
    int k0 = ks * 32;
    ABReg A, Bh, Bt, Be;
    const u32x4* ap = (const u32x4*)(xbf + (m0 + l15) * F + k0 + asel * 16);
    A.q[0] = ap[0];
    A.q[1] = ap[1];                       // +8 halfs
    int boff = (colbase + l15) * F + k0 + asel * 16;
    const u32x4* bp;
    bp = (const u32x4*)(wh + boff); Bh.q[0] = bp[0]; Bh.q[1] = bp[1];
    bp = (const u32x4*)(wt + boff); Bt.q[0] = bp[0]; Bt.q[1] = bp[1];
    bp = (const u32x4*)(we + boff); Be.q[0] = bp[0]; Be.q[1] = bp[1];
    a0 = __builtin_amdgcn_wmma_f32_16x16x32_bf16(false, A.v, false, Bh.v, (short)0, a0, false, false);
    a1 = __builtin_amdgcn_wmma_f32_16x16x32_bf16(false, A.v, false, Bt.v, (short)0, a1, false, false);
    a2 = __builtin_amdgcn_wmma_f32_16x16x32_bf16(false, A.v, false, Be.v, (short)0, a2, false, false);
  }
  int col = colbase + l15;
#pragma unroll
  for (int r = 0; r < 8; r++) {
    int row = m0 + asel * 8 + r;
    float s = hn[row];
    fh[row * F + col] = a0[r] * s;
    ft[row * F + col] = a1[r] * s;
    fe[row * F + col] = a2[r] * s;
  }
}

// ---------------- edge attention logits + segment max ----------------
__global__ void k_attn(const int* __restrict__ src, const int* __restrict__ dst,
                       const float* __restrict__ fh, const float* __restrict__ ft,
                       const float* __restrict__ attn, const float* __restrict__ li,
                       float* __restrict__ ae, unsigned* __restrict__ nmax) {
  int e = blockIdx.x * blockDim.x + threadIdx.x;
  if (e >= EE) return;
  int s = src[e], d = dst[e];
  const float4* hp = (const float4*)(fh + s * F);
  const float4* tp = (const float4*)(ft + d * F);
  const float4* wp = (const float4*)attn;
  float scale = li[d] * (1.0f / DD);
#pragma unroll
  for (int h = 0; h < HH; h++) {
    float acc = 0.f;
#pragma unroll
    for (int q = 0; q < 4; q++) {
      float4 a4 = hp[h * 4 + q], b4 = tp[h * 4 + q], w4 = wp[h * 4 + q];
      acc += lrelu(a4.x * b4.x) * w4.x;
      acc += lrelu(a4.y * b4.y) * w4.y;
      acc += lrelu(a4.z * b4.z) * w4.z;
      acc += lrelu(a4.w * b4.w) * w4.w;
    }
    float ev = acc * scale;
    ae[e * HH + h] = ev;
    atomicMax(nmax + d * HH + h, fkey(ev));
  }
}

__global__ void k_exp(const int* __restrict__ dst, const unsigned* __restrict__ nmax,
                      float* __restrict__ ae, float* __restrict__ den) {
  int e = blockIdx.x * blockDim.x + threadIdx.x;
  if (e >= EE) return;
  int d = dst[e];
#pragma unroll
  for (int h = 0; h < HH; h++) {
    float m = fdec(nmax[d * HH + h]);
    float v = expf(ae[e * HH + h] - m);
    ae[e * HH + h] = v;
    atomicAdd(den + d * HH + h, v);
  }
}

__global__ void k_anorm(const int* __restrict__ dst, float* __restrict__ ae,
                        const float* __restrict__ den) {
  int i = blockIdx.x * blockDim.x + threadIdx.x;
  if (i >= EE * HH) return;
  int e = i >> 3, h = i & 7;
  ae[i] = ae[i] / den[dst[e] * HH + h];
}

// ---------------- PPR diffusion: scatter + finalize ----------------
__global__ void k_scatter(const int* __restrict__ src, const int* __restrict__ dst,
                          const float* __restrict__ ae, const float* __restrict__ hcur,
                          const float* __restrict__ hn, float* __restrict__ hout,
                          int useScale) {
  int i = blockIdx.x * blockDim.x + threadIdx.x;
  if (i >= EE * HH) return;
  int e = i >> 3, h = i & 7;
  int s = src[e], d = dst[e];
  float w = ae[i];
  if (useScale) w *= hn[s];     // h = h * head_norm for hop > 0 (applied at gather)
  const float4* hp = (const float4*)(hcur + s * F + h * DD);
  float* op = hout + d * F + h * DD;
#pragma unroll
  for (int q = 0; q < 4; q++) {
    float4 v = hp[q];
    atomicAdd(op + q * 4 + 0, v.x * w);
    atomicAdd(op + q * 4 + 1, v.y * w);
    atomicAdd(op + q * 4 + 2, v.z * w);
    atomicAdd(op + q * 4 + 3, v.w * w);
  }
}

__global__ void k_final(float* __restrict__ h, const float* __restrict__ fe,
                        const float* __restrict__ tn) {
  int i = blockIdx.x * blockDim.x + threadIdx.x;
  if (i >= NN * F) return;
  int n = i >> 7;
  h[i] = (1.0f - ALPHA) * (h[i] * tn[n]) + ALPHA * fe[i];
}

// ---------------- residual + LN2 (store rst f32 and bf16 LN out) ----------------
__global__ void k_rst_ln(const float* __restrict__ hfin, const float* __restrict__ feat,
                         const float* __restrict__ g, const float* __restrict__ b,
                         float* __restrict__ rst, unsigned short* __restrict__ x2) {
  int wave = threadIdx.x >> 5, lane = threadIdx.x & 31;
  int row = blockIdx.x * 8 + wave;
  float4 hv = ((const float4*)(hfin + row * F))[lane];
  float4 fv = ((const float4*)(feat + row * F))[lane];
  float4 r;
  r.x = hv.x + fv.x; r.y = hv.y + fv.y; r.z = hv.z + fv.z; r.w = hv.w + fv.w;
  ((float4*)(rst + row * F))[lane] = r;
  float s = r.x + r.y + r.z + r.w;
  for (int o = 16; o > 0; o >>= 1) s += __shfl_xor(s, o, 32);
  float mu = s * (1.0f / F);
  float dx = r.x - mu, dy = r.y - mu, dz = r.z - mu, dw = r.w - mu;
  float q = dx * dx + dy * dy + dz * dz + dw * dw;
  for (int o = 16; o > 0; o >>= 1) q += __shfl_xor(q, o, 32);
  float rs = rsqrtf(q * (1.0f / F) + LNEPS);
  float4 g4 = ((const float4*)g)[lane];
  float4 b4 = ((const float4*)b)[lane];
  float y0 = dx * rs * g4.x + b4.x;
  float y1 = dy * rs * g4.y + b4.y;
  float y2 = dz * rs * g4.z + b4.z;
  float y3 = dw * rs * g4.w + b4.w;
  uint2 pk;
  pk.x = (unsigned)f2bf(y0) | ((unsigned)f2bf(y1) << 16);
  pk.y = (unsigned)f2bf(y2) | ((unsigned)f2bf(y3) << 16);
  *(uint2*)(x2 + row * F + lane * 4) = pk;
}

// ---------------- FFN layer 1: y1 = relu(x2 @ W1^T + b1), bf16 out ----------------
__global__ void __launch_bounds__(256) k_ffn1(
    const unsigned short* __restrict__ x2, const unsigned short* __restrict__ w1,
    const float* __restrict__ b1, unsigned short* __restrict__ y1) {
  int wave = threadIdx.x >> 5, lane = threadIdx.x & 31;
  int l15 = lane & 15, asel = lane >> 4;
  int m0 = blockIdx.x * 16;
  v8f z = {0,0,0,0,0,0,0,0};
  v8f acc[4] = {z, z, z, z};
#pragma unroll
  for (int ks = 0; ks < 4; ks++) {
    int k0 = ks * 32;
    ABReg A;
    const u32x4* ap = (const u32x4*)(x2 + (m0 + l15) * F + k0 + asel * 16);
    A.q[0] = ap[0];
    A.q[1] = ap[1];
#pragma unroll
    for (int j = 0; j < 4; j++) {
      int cb = (wave * 4 + j) * 16;
      ABReg B;
      const u32x4* bp = (const u32x4*)(w1 + (cb + l15) * F + k0 + asel * 16);
      B.q[0] = bp[0]; B.q[1] = bp[1];
      acc[j] = __builtin_amdgcn_wmma_f32_16x16x32_bf16(false, A.v, false, B.v, (short)0, acc[j], false, false);
    }
  }
#pragma unroll
  for (int j = 0; j < 4; j++) {
    int col = (wave * 4 + j) * 16 + l15;
    float bias = b1[col];
#pragma unroll
    for (int r = 0; r < 8; r++) {
      int row = m0 + asel * 8 + r;
      float v = fmaxf(acc[j][r] + bias, 0.0f);
      y1[row * FFH + col] = f2bf(v);
    }
  }
}

// ---------------- FFN layer 2 + residual: out = y1 @ W2^T + b2 + rst ----------------
__global__ void __launch_bounds__(256) k_ffn2(
    const unsigned short* __restrict__ y1, const unsigned short* __restrict__ w2,
    const float* __restrict__ b2, const float* __restrict__ rst, float* __restrict__ out) {
  int wave = threadIdx.x >> 5, lane = threadIdx.x & 31;
  int l15 = lane & 15, asel = lane >> 4;
  int m0 = blockIdx.x * 16;
  int colbase = wave * 16;
  v8f acc = {0,0,0,0,0,0,0,0};
#pragma unroll
  for (int ks = 0; ks < 16; ks++) {
    int k0 = ks * 32;
    ABReg A, B;
    const u32x4* ap = (const u32x4*)(y1 + (m0 + l15) * FFH + k0 + asel * 16);
    A.q[0] = ap[0];
    A.q[1] = ap[1];
    const u32x4* bp = (const u32x4*)(w2 + (colbase + l15) * FFH + k0 + asel * 16);
    B.q[0] = bp[0]; B.q[1] = bp[1];
    acc = __builtin_amdgcn_wmma_f32_16x16x32_bf16(false, A.v, false, B.v, (short)0, acc, false, false);
  }
  int col = colbase + l15;
  float bias = b2[col];
#pragma unroll
  for (int r = 0; r < 8; r++) {
    int row = m0 + asel * 8 + r;
    out[row * F + col] = acc[r] + bias + rst[row * F + col];
  }
}

// ---------------- host launcher ----------------
extern "C" void kernel_launch(void* const* d_in, const int* in_sizes, int n_in,
                              void* d_out, int out_size, void* d_ws, size_t ws_size,
                              hipStream_t stream) {
  (void)in_sizes; (void)n_in; (void)out_size; (void)ws_size;
  const float* feat  = (const float*)d_in[0];
  const int*   src   = (const int*)d_in[1];
  const int*   dst   = (const int*)d_in[2];
  const float* w_head = (const float*)d_in[3];
  const float* w_tail = (const float*)d_in[4];
  const float* w_ent  = (const float*)d_in[5];
  const float* attn   = (const float*)d_in[6];
  const float* ln1_g = (const float*)d_in[7];
  const float* ln1_b = (const float*)d_in[8];
  const float* ln2_g = (const float*)d_in[9];
  const float* ln2_b = (const float*)d_in[10];
  const float* ff_w1 = (const float*)d_in[11];
  const float* ff_b1 = (const float*)d_in[12];
  const float* ff_w2 = (const float*)d_in[13];
  const float* ff_b2 = (const float*)d_in[14];
  float* out = (float*)d_out;

  char* ws = (char*)d_ws;
  size_t o = 0;
  auto alloc = [&](size_t bytes) -> char* {
    char* p = ws + o;
    o += (bytes + 255) & ~(size_t)255;
    return p;
  };
  unsigned short* xbf = (unsigned short*)alloc((size_t)NN * F * 2);  // aliased as x2 later
  float* fh = (float*)alloc((size_t)NN * F * 4);                     // aliased as rst later
  float* ft = (float*)alloc((size_t)NN * F * 4);                     // aliased as y1 (part 1)
  float* fe = (float*)alloc((size_t)NN * F * 4);                     //  ...   y1 (part 2)
  float* hA = (float*)alloc((size_t)NN * F * 4);
  float* hB = (float*)alloc((size_t)NN * F * 4);
  float* ae = (float*)alloc((size_t)EE * HH * 4);
  float* odeg = (float*)alloc((size_t)NN * 4);
  float* ideg = (float*)alloc((size_t)NN * 4);
  float* hn   = (float*)alloc((size_t)NN * 4);
  float* tn   = (float*)alloc((size_t)NN * 4);
  float* li   = (float*)alloc((size_t)NN * 4);
  unsigned* nmax = (unsigned*)alloc((size_t)NN * HH * 4);
  float* den     = (float*)alloc((size_t)NN * HH * 4);
  unsigned short* whb = (unsigned short*)alloc((size_t)F * F * 2);
  unsigned short* wtb = (unsigned short*)alloc((size_t)F * F * 2);
  unsigned short* web = (unsigned short*)alloc((size_t)F * F * 2);
  unsigned short* w1b = (unsigned short*)alloc((size_t)FFH * F * 2);
  unsigned short* w2b = (unsigned short*)alloc((size_t)F * FFH * 2);
  // aliases (lifetimes do not overlap)
  float* rst = fh;
  unsigned short* x2 = xbf;
  unsigned short* y1 = (unsigned short*)ft;   // ft+fe contiguous = NN*512*2 bytes

  // --- degrees & norms ---
  k_fill<<<512, 256, 0, stream>>>(odeg, NN, 0.f);
  k_fill<<<512, 256, 0, stream>>>(ideg, NN, 0.f);
  k_fill<<<512, 256, 0, stream>>>((float*)nmax, NN * HH, 0.f);  // key 0 == -inf sentinel
  k_fill<<<512, 256, 0, stream>>>(den, NN * HH, 0.f);
  k_deg<<<EE / 256, 256, 0, stream>>>(src, dst, odeg, ideg);
  k_norms<<<(NN + 255) / 256, 256, 0, stream>>>(odeg, ideg, hn, tn, li);

  // --- weight conversion to bf16 ---
  k_cvt_bf16<<<64, 256, 0, stream>>>(w_head, whb, F * F);
  k_cvt_bf16<<<64, 256, 0, stream>>>(w_tail, wtb, F * F);
  k_cvt_bf16<<<64, 256, 0, stream>>>(w_ent,  web, F * F);
  k_cvt_bf16<<<256, 256, 0, stream>>>(ff_w1, w1b, FFH * F);
  k_cvt_bf16<<<256, 256, 0, stream>>>(ff_w2, w2b, F * FFH);

  // --- LN1 + triple projection (WMMA) ---
  k_ln_cast<<<NN / 8, 256, 0, stream>>>(feat, ln1_g, ln1_b, xbf);
  k_gemm3<<<NN / 16, 256, 0, stream>>>(xbf, whb, wtb, web, hn, fh, ft, fe);

  // --- edge softmax ---
  k_attn<<<EE / 256, 256, 0, stream>>>(src, dst, fh, ft, attn, li, ae, nmax);
  k_exp<<<EE / 256, 256, 0, stream>>>(dst, nmax, ae, den);
  k_anorm<<<EE * HH / 256, 256, 0, stream>>>(dst, ae, den);

  // --- 5-hop PPR diffusion (ping-pong hB/hA, final lands in hB) ---
  const float* cur = fe;
  float* pp[2] = {hB, hA};
  for (int hop = 0; hop < NHOP; hop++) {
    float* ho = pp[hop & 1];
    k_fill<<<4096, 256, 0, stream>>>(ho, NN * F, 0.f);
    k_scatter<<<EE * HH / 256, 256, 0, stream>>>(src, dst, ae, cur, hn, ho, hop > 0 ? 1 : 0);
    k_final<<<NN * F / 256, 256, 0, stream>>>(ho, fe, tn);
    cur = ho;
  }

  // --- residual + LN2 + FFN (WMMA) + residual ---
  k_rst_ln<<<NN / 8, 256, 0, stream>>>(cur, feat, ln2_g, ln2_b, rst, x2);
  k_ffn1<<<NN / 16, 256, 0, stream>>>(x2, w1b, ff_b1, y1);
  k_ffn2<<<NN / 16, 256, 0, stream>>>(y1, w2b, ff_b2, rst, out);
}
